// ICLAttention_7335804142256
// MI455X (gfx1250) — matrix operand order repaired and checked by reading, not compiled, not run on hardware
//
#include <hip/hip_runtime.h>

// Problem constants (from reference)
#define BB_ 2
#define SS_ 2048
#define EE_ 1024
#define HH_ 16
#define DH_ 64
#define SCALE_ (1.0f / 32.0f)                       // 1/sqrt(E), per reference
#define QSCALE_ (SCALE_ * 1.44269504088896340736f)  // fold log2(e): softmax in base-2 domain

typedef __attribute__((ext_vector_type(16))) __bf16 bf16x16;
typedef __attribute__((ext_vector_type(2)))  __bf16 bf16x2;
typedef __attribute__((ext_vector_type(8)))  float  f32x8;
typedef __attribute__((ext_vector_type(4)))  unsigned int u32x4;
typedef __attribute__((ext_vector_type(8)))  int          i32x8;
typedef __attribute__((ext_vector_type(4)))  int          i32x4;

#if __has_builtin(__builtin_amdgcn_tensor_load_to_lds) && __has_builtin(__builtin_amdgcn_s_wait_tensorcnt)
#define USE_TDM 1
#else
#define USE_TDM 0
#endif

union BFrag {
    bf16x16        v;
    unsigned int   p[8];
    unsigned short s[16];
};

__device__ __forceinline__ unsigned short f2bf(float f) {
    unsigned int u = __float_as_uint(f);
    u += 0x7FFFu + ((u >> 16) & 1u);       // round to nearest even
    return (unsigned short)(u >> 16);
}
__device__ __forceinline__ float bf2f(unsigned short h) {
    return __uint_as_float(((unsigned int)h) << 16);
}
// two f32 -> packed bf16: single v_cvt_pk if present, else 3 VALU ops
// (round-half-up adds + one v_perm_b32 byte select), else manual RNE.
__device__ __forceinline__ unsigned pack_bf16(float a, float b) {
#if __has_builtin(__builtin_amdgcn_cvt_pk_bf16_f32)
    union { bf16x2 v; unsigned u; } cv;
    cv.v = __builtin_amdgcn_cvt_pk_bf16_f32(a, b);
    return cv.u;
#elif __has_builtin(__builtin_amdgcn_perm)
    unsigned ua = __float_as_uint(a) + 0x8000u;
    unsigned ub = __float_as_uint(b) + 0x8000u;
    return __builtin_amdgcn_perm(ua, ub, 0x03020706u);  // {ub[31:16], ua[31:16]}
#else
    return (unsigned)f2bf(a) | ((unsigned)f2bf(b) << 16);
#endif
}
// pack the low/high 16-bit halves of two dwords (V pair staging)
__device__ __forceinline__ unsigned perm_lo16(unsigned adw, unsigned bdw) {
#if __has_builtin(__builtin_amdgcn_perm)
    return __builtin_amdgcn_perm(bdw, adw, 0x05040100u);  // {b[15:0], a[15:0]}
#else
    return (adw & 0xFFFFu) | (bdw << 16);
#endif
}
__device__ __forceinline__ unsigned perm_hi16(unsigned adw, unsigned bdw) {
#if __has_builtin(__builtin_amdgcn_perm)
    return __builtin_amdgcn_perm(bdw, adw, 0x07060302u);  // {b[31:16], a[31:16]}
#else
    return (adw >> 16) | (bdw & 0xFFFF0000u);
#endif
}
__device__ __forceinline__ float fexp2(float x) {
#if __has_builtin(__builtin_amdgcn_exp2f)
    return __builtin_amdgcn_exp2f(x);
#else
    return exp2f(x);
#endif
}

// cross-half (lane ^ 16) exchange: v_permlanex16_b32, pure VALU
__device__ __forceinline__ float permx16(float x) {
#if __has_builtin(__builtin_amdgcn_permlanex16)
    unsigned u = __float_as_uint(x);
    unsigned r = __builtin_amdgcn_permlanex16(u, u, 0x76543210u, 0xFEDCBA98u, false, false);
    return __uint_as_float(r);
#else
    return __shfl_xor(x, 16, 32);
#endif
}

// A-matrix 16x32 fragment (row per lane, contiguous K pairs)
__device__ __forceinline__ bf16x16 load_frag_a(const unsigned short* rowbase, int half) {
    BFrag fr;
#pragma unroll
    for (int v = 0; v < 8; ++v) {
        int k0 = half * 16 + v * 2;
        fr.p[v] = *(const unsigned int*)(rowbase + k0);
    }
    return fr.v;
}
// B-matrix 32x16 fragment (column per lane, contiguous K pairs)
__device__ __forceinline__ bf16x16 load_frag_b(const unsigned short* colbase, int half) {
    BFrag fr;
#pragma unroll
    for (int v = 0; v < 8; ++v) {
        int k0 = half * 16 + v * 2;
        fr.p[v] = *(const unsigned int*)(colbase + k0);
    }
    return fr.v;
}

__device__ __forceinline__ f32x8 wmma_bf16(bf16x16 a, bf16x16 b, f32x8 c) {
    return __builtin_amdgcn_wmma_f32_16x16x32_bf16(false, a, false, b, (short)0, c, false, false);
}

// ---------------------------------------------------------------- convert f32 -> bf16 (x4)
__global__ void cvt_bf16_kernel(const float4* __restrict__ in, uint2* __restrict__ out, int n4) {
    int i = blockIdx.x * blockDim.x + threadIdx.x;
    if (i < n4) {
        float4 f = in[i];
        uint2 r;
        r.x = pack_bf16(f.x, f.y);
        r.y = pack_bf16(f.z, f.w);
        out[i] = r;
    }
}

// ---------------------------------------------------------------- RoPE (in place, bf16)
__global__ void rope_kernel(unsigned short* __restrict__ X, int n_pairs, float post_scale) {
    int idx = blockIdx.x * blockDim.x + threadIdx.x;
    if (idx >= n_pairs) return;
    int cp  = idx & (EE_ / 2 - 1);
    int row = idx >> 9;                 // idx / (E/2)
    int s   = row & (SS_ - 1);          // position
    int h   = cp >> 5;                  // / (Dh/2)
    int i   = cp & 31;
    float inv_freq = __powf(10000.0f, -(float)(2 * i) / (float)DH_);
    float ang = (float)s * inv_freq;
    float c, sn;
    __sincosf(ang, &sn, &c);
    unsigned* p = (unsigned*)(X + (size_t)row * EE_ + h * DH_ + 2 * i);
    unsigned w = *p;
    float xe = bf2f((unsigned short)(w & 0xFFFF));
    float xo = bf2f((unsigned short)(w >> 16));
    *p = pack_bf16((xe * c - xo * sn) * post_scale, (xo * c + xe * sn) * post_scale);
}

// ---------------------------------------------------------------- bf16 WMMA GEMM
template <bool OUT_BF16>
__global__ void gemm_bf16_wmma(const unsigned short* __restrict__ A,
                               const unsigned short* __restrict__ W,
                               void* __restrict__ Out,
                               int M, int N, int K) {
    __shared__ __align__(16) unsigned short sA[64 * 32];
    __shared__ __align__(16) unsigned short sW[64 * 32];

    const int t     = threadIdx.x;          // 0..127
    const int wave  = t >> 5;
    const int lane  = t & 31;
    const int half  = lane >> 4;
    const int ln    = lane & 15;
    const int waveM = wave >> 1;
    const int waveN = wave & 1;
    const int mBase = blockIdx.y * 64;
    const int nBase = blockIdx.x * 64;

    const int ldr = t >> 1;
    const int ldc = (t & 1) * 16;

    f32x8 acc[2][2] = {};                   // acc[tm][tn]: rows=n (VGPR), cols=m (lane)

    for (int kb = 0; kb < K; kb += 32) {
        {
            const uint4* ga = (const uint4*)(A + (size_t)(mBase + ldr) * K + kb + ldc);
            uint4* pa = (uint4*)(sA + ldr * 32 + ldc);
            pa[0] = ga[0]; pa[1] = ga[1];
            const uint4* gw = (const uint4*)(W + (size_t)(nBase + ldr) * K + kb + ldc);
            uint4* pw = (uint4*)(sW + ldr * 32 + ldc);
            pw[0] = gw[0]; pw[1] = gw[1];
        }
        __syncthreads();

        bf16x16 aw[2], bx[2];
#pragma unroll
        for (int tn = 0; tn < 2; ++tn)
            aw[tn] = load_frag_a(sW + (waveN * 32 + tn * 16 + ln) * 32, half);
#pragma unroll
        for (int tm = 0; tm < 2; ++tm)
            bx[tm] = load_frag_b(sA + (waveM * 32 + tm * 16 + ln) * 32, half);

#pragma unroll
        for (int tm = 0; tm < 2; ++tm)
#pragma unroll
            for (int tn = 0; tn < 2; ++tn)
                acc[tm][tn] = wmma_bf16(aw[tn], bx[tm], acc[tm][tn]);

        __syncthreads();
    }

#pragma unroll
    for (int tm = 0; tm < 2; ++tm) {
        int row = mBase + waveM * 32 + tm * 16 + ln;
#pragma unroll
        for (int tn = 0; tn < 2; ++tn) {
            int colB = nBase + waveN * 32 + tn * 16 + half * 8;
            if (OUT_BF16) {
                unsigned short* op = (unsigned short*)Out + (size_t)row * N + colB;
#pragma unroll
                for (int r = 0; r < 8; r += 2)
                    *(unsigned*)(op + r) = pack_bf16(acc[tm][tn][r], acc[tm][tn][r + 1]);
            } else {
                float* op = (float*)Out + (size_t)row * N + colB;
                float4 f0 = make_float4(acc[tm][tn][0], acc[tm][tn][1], acc[tm][tn][2], acc[tm][tn][3]);
                float4 f1 = make_float4(acc[tm][tn][4], acc[tm][tn][5], acc[tm][tn][6], acc[tm][tn][7]);
                *(float4*)(op)     = f0;
                *(float4*)(op + 4) = f1;
            }
        }
    }
}

// ---------------------------------------------------------------- flash attention (bf16 WMMA)
// Key tile = 64. K tile staged by the Tensor Data Mover (one DMA per tile, issued
// by wave 0, tracked with TENSORcnt); V staged manually in permuted-transposed
// layout (TDM cannot transpose).  Scores transposed, base-2 softmax, P fragments
// built in registers.
__global__ void attn_wmma_kernel(const unsigned short* __restrict__ Qp,
                                 const unsigned short* __restrict__ Kp,
                                 const unsigned short* __restrict__ Vp,
                                 unsigned short* __restrict__ Op) {
    __shared__ __align__(16) unsigned short sK[64 * 64];    // [key][d]      8 KB
    __shared__ __align__(16) unsigned short sVt[64 * 64];   // [d][k']       8 KB

    const int t    = threadIdx.x;
    const int wave = t >> 5;
    const int lane = t & 31;
    const int half = lane >> 4;
    const int ln   = lane & 15;
    const int bb   = blockIdx.z;
    const int h    = blockIdx.y;
    const int hcol = h * DH_;
    const int qTile = blockIdx.x * 64;

    const unsigned short* qbase =
        Qp + (size_t)(bb * SS_ + qTile + wave * 16 + ln) * EE_ + hcol;
    bf16x16 bq0 = load_frag_b(qbase, half);
    bf16x16 bq1 = load_frag_b(qbase + 32, half);

    f32x8 o[4] = {};            // O^T: o[tn][r] = O[d = tn*16 + half*8 + r][query ln]
    float mrun = -3.0e38f, lrun = 0.0f;

#if !USE_TDM
    const int kr = t >> 1;                  // 0..63
    const int kc = (t & 1) * 32;            // 0 or 32
#endif
    // V staging: thread handles key pair (2*pi, 2*pi+1), 16 d-cols at pd
    const int pi = t & 31;                  // pair 0..31
    const int pd = (t >> 5) * 16;           // d offset 0..48
    const int ka = 2 * pi;
    const int kp = ((ka >> 5) << 5) + (((ka >> 3) & 1) << 4) +
                   (((ka >> 4) & 1) << 3) + (ka & 7);        // permuted key

    for (int kt = 0; kt < SS_ / 64; ++kt) {
        const int keyBase = kt * 64;
#if USE_TDM
        if (t < 32) {   // wave 0: one TDM DMA moves the whole 64x64 bf16 K tile
            const unsigned short* gkt = Kp + (size_t)(bb * SS_ + keyBase) * EE_ + hcol;
            unsigned long long ga = (unsigned long long)(size_t)gkt;
            u32x4 g0;
            g0[0] = 1u;                                   // count=1, user mode
            g0[1] = (unsigned)(size_t)(&sK[0]);           // lds_addr
            g0[2] = (unsigned)ga;                         // global_addr[31:0]
            g0[3] = (unsigned)(ga >> 32) | 0x80000000u;   // addr[56:32] | type=2
            i32x8 g1;
            g1[0] = 0x00010000;        // workgroup_mask=0, data_size=2B
            g1[1] = (int)(64u << 16);  // tensor_dim0=64 (lo16 at bit48)
            g1[2] = (int)(64u << 16);  // tensor_dim0 hi | tensor_dim1=64 lo16
            g1[3] = (int)(64u << 16);  // tensor_dim1 hi | tile_dim0=64
            g1[4] = 64;                // tile_dim1=64, tile_dim2=0
            g1[5] = EE_;               // tensor_dim0_stride = 1024 elements
            g1[6] = 0;                 // stride hi | tensor_dim1_stride lo
            g1[7] = 0;
            i32x4 gz4 = {0, 0, 0, 0};
            i32x8 gz8 = {0, 0, 0, 0, 0, 0, 0, 0};
            __builtin_amdgcn_tensor_load_to_lds(g0, g1, gz4, gz4, gz8, 0);
            __builtin_amdgcn_s_wait_tensorcnt(0);
        }
#else
        const unsigned short* gk = Kp + (size_t)(bb * SS_ + keyBase + kr) * EE_ + hcol + kc;
        {
            uint4* pk = (uint4*)(sK + kr * 64 + kc);
            pk[0] = ((const uint4*)gk)[0];
            pk[1] = ((const uint4*)gk)[1];
            pk[2] = ((const uint4*)gk)[2];
            pk[3] = ((const uint4*)gk)[3];
        }
        if (kt + 1 < SS_ / 64) __builtin_prefetch(gk + 64 * EE_, 0, 1);
#endif
        const unsigned short* gv0 = Vp + (size_t)(bb * SS_ + keyBase + ka) * EE_ + hcol + pd;
        const unsigned short* gv1 = gv0 + EE_;
        {
            union { uint4 q[2]; unsigned u[8]; } a, b;
            a.q[0] = ((const uint4*)gv0)[0];
            a.q[1] = ((const uint4*)gv0)[1];
            b.q[0] = ((const uint4*)gv1)[0];
            b.q[1] = ((const uint4*)gv1)[1];
#pragma unroll
            for (int cw = 0; cw < 8; ++cw) {   // 2 byte-perms pack each dword pair
                *(unsigned*)&sVt[(pd + 2 * cw) * 64 + kp]     = perm_lo16(a.u[cw], b.u[cw]);
                *(unsigned*)&sVt[(pd + 2 * cw + 1) * 64 + kp] = perm_hi16(a.u[cw], b.u[cw]);
            }
        }
        if (kt + 1 < SS_ / 64) {
            __builtin_prefetch(gv0 + 64 * EE_, 0, 1);
            __builtin_prefetch(gv1 + 64 * EE_, 0, 1);
        }
        __syncthreads();

        // S^T tiles: A = K rows (16 keys x 64 d), B = Q^T (Q pre-scaled by log2e/sqrt(E))
        f32x8 sc[4];
#pragma unroll
        for (int tt = 0; tt < 4; ++tt) {
            const unsigned short* kb0 = sK + (tt * 16 + ln) * 64;
            bf16x16 ak0 = load_frag_a(kb0, half);
            bf16x16 ak1 = load_frag_a(kb0 + 32, half);
            f32x8 c = {};
            c = wmma_bf16(ak0, bq0, c);
            c = wmma_bf16(ak1, bq1, c);
            sc[tt] = c;
        }

        // per-lane online softmax (base-2) over 32 key scores + one cross-half merge
        float ml = -3.0e38f;
#pragma unroll
        for (int tt = 0; tt < 4; ++tt) {
            float m = fmaxf(fmaxf(fmaxf(sc[tt][0], sc[tt][1]), fmaxf(sc[tt][2], sc[tt][3])),
                            fmaxf(fmaxf(sc[tt][4], sc[tt][5]), fmaxf(sc[tt][6], sc[tt][7])));
            ml = fmaxf(ml, m);
        }
        float mt = fmaxf(ml, permx16(ml));
        float mn = fmaxf(mrun, mt);
        float alpha = fexp2(mrun - mn);
        mrun = mn;

        float ls = 0.0f;
#pragma unroll
        for (int tt = 0; tt < 4; ++tt)
#pragma unroll
            for (int r = 0; r < 8; ++r) {
                float p = fexp2(sc[tt][r] - mn);
                sc[tt][r] = p;
                ls += p;
            }
        BFrag pb[2];
#pragma unroll
        for (int g = 0; g < 2; ++g)
#pragma unroll
            for (int t2 = 0; t2 < 2; ++t2)
#pragma unroll
                for (int r = 0; r < 8; r += 2)
                    pb[g].p[t2 * 4 + r / 2] =
                        pack_bf16(sc[2 * g + t2][r], sc[2 * g + t2][r + 1]);
        float lt = ls + permx16(ls);
        lrun = lrun * alpha + lt;
#pragma unroll
        for (int tn = 0; tn < 4; ++tn)
#pragma unroll
            for (int r = 0; r < 8; ++r) o[tn][r] *= alpha;

        // O^T += V^T (16 d x 64 k') @ P (64 k' x 16 q): 8 WMMAs, P straight from regs
#pragma unroll
        for (int tn = 0; tn < 4; ++tn) {
            const unsigned short* vrow = sVt + (tn * 16 + ln) * 64;
            bf16x16 av0 = load_frag_a(vrow, half);
            bf16x16 av1 = load_frag_a(vrow + 32, half);
            o[tn] = wmma_bf16(av0, pb[0].v, o[tn]);
            o[tn] = wmma_bf16(av1, pb[1].v, o[tn]);
        }

        __syncthreads();
    }

    // normalize + store: lane's 8 values per tile are contiguous in d -> packed b32
    float inv = 1.0f / lrun;
    unsigned short* op = Op + (size_t)(bb * SS_ + qTile + wave * 16 + ln) * EE_ + hcol + half * 8;
#pragma unroll
    for (int tn = 0; tn < 4; ++tn)
#pragma unroll
        for (int r = 0; r < 8; r += 2)
            *(unsigned*)(op + tn * 16 + r) = pack_bf16(o[tn][r] * inv, o[tn][r + 1] * inv);
}

// ---------------------------------------------------------------- launch
extern "C" void kernel_launch(void* const* d_in, const int* in_sizes, int n_in,
                              void* d_out, int out_size, void* d_ws, size_t ws_size,
                              hipStream_t stream) {
    const float* q   = (const float*)d_in[0];
    const float* k   = (const float*)d_in[1];
    const float* v   = (const float*)d_in[2];
    const float* W_q = (const float*)d_in[3];
    const float* W_k = (const float*)d_in[4];
    const float* W_v = (const float*)d_in[5];
    const float* W_o = (const float*)d_in[6];
    float* out = (float*)d_out;

    const int M  = BB_ * SS_;            // 4096
    const int nX = M * EE_;
    const int nW = EE_ * EE_;

    unsigned char* w = (unsigned char*)d_ws;
    auto take = [&](size_t bytes) { unsigned short* p = (unsigned short*)w; w += bytes; return p; };
    const size_t xb = (size_t)nX * 2;    // 8 MB
    const size_t wb = (size_t)nW * 2;    // 2 MB
    unsigned short* xq = take(xb);
    unsigned short* xk = take(xb);
    unsigned short* xv = take(xb);
    unsigned short* wq = take(wb);
    unsigned short* wk = take(wb);
    unsigned short* wv = take(wb);
    unsigned short* wo = take(wb);
    unsigned short* Qp = take(xb);
    unsigned short* Kp = take(xb);
    unsigned short* Vp = take(xb);
    unsigned short* Ao = take(xb);       // total ~64 MB

    const int nX4 = nX / 4, nW4 = nW / 4;
    cvt_bf16_kernel<<<(nX4 + 255) / 256, 256, 0, stream>>>((const float4*)q, (uint2*)xq, nX4);
    cvt_bf16_kernel<<<(nX4 + 255) / 256, 256, 0, stream>>>((const float4*)k, (uint2*)xk, nX4);
    cvt_bf16_kernel<<<(nX4 + 255) / 256, 256, 0, stream>>>((const float4*)v, (uint2*)xv, nX4);
    cvt_bf16_kernel<<<(nW4 + 255) / 256, 256, 0, stream>>>((const float4*)W_q, (uint2*)wq, nW4);
    cvt_bf16_kernel<<<(nW4 + 255) / 256, 256, 0, stream>>>((const float4*)W_k, (uint2*)wk, nW4);
    cvt_bf16_kernel<<<(nW4 + 255) / 256, 256, 0, stream>>>((const float4*)W_v, (uint2*)wv, nW4);
    cvt_bf16_kernel<<<(nW4 + 255) / 256, 256, 0, stream>>>((const float4*)W_o, (uint2*)wo, nW4);

    dim3 gblk(128);
    dim3 ggrd(EE_ / 64, M / 64);
    gemm_bf16_wmma<true><<<ggrd, gblk, 0, stream>>>(xq, wq, Qp, M, EE_, EE_);
    gemm_bf16_wmma<true><<<ggrd, gblk, 0, stream>>>(xk, wk, Kp, M, EE_, EE_);
    gemm_bf16_wmma<true><<<ggrd, gblk, 0, stream>>>(xv, wv, Vp, M, EE_, EE_);

    const int npairs = nX / 2;
    rope_kernel<<<(npairs + 255) / 256, 256, 0, stream>>>(Qp, npairs, QSCALE_);  // scale & log2e into Q
    rope_kernel<<<(npairs + 255) / 256, 256, 0, stream>>>(Kp, npairs, 1.0f);

    dim3 agrd(SS_ / 64, HH_, BB_);
    attn_wmma_kernel<<<agrd, dim3(128), 0, stream>>>(Qp, Kp, Vp, Ao);

    gemm_bf16_wmma<false><<<ggrd, gblk, 0, stream>>>(Ao, wo, out, M, EE_, EE_);
}
